// GNNet_5420248728073
// MI455X (gfx1250) — matrix operand order repaired and checked by reading, not compiled, not run on hardware
//
#include <hip/hip_runtime.h>

#define N_NODES 1500
#define N_EDGES 24000
#define EMB     64
#define LOOP_N  10   // reference constant LOOP

typedef __attribute__((ext_vector_type(16))) _Float16 v16h;
typedef __attribute__((ext_vector_type(8)))  _Float16 v8h;
typedef __attribute__((ext_vector_type(8)))  float    v8f;

// ---------------------------------------------------------------- utilities
__device__ __forceinline__ void atomicMaxF(float* addr, float val) {
  // sign-aware float max via integer atomics (works with -inf init)
  if (val >= 0.0f) atomicMax((int*)addr, __float_as_int(val));
  else             atomicMin((unsigned int*)addr, __float_as_uint(val));
}

// ------------------------------------------------------------- preparation
// vc[i] = concat(v[i,0:7], labels[i])
__global__ void k_vc(const float* v, const float* labels, float* vc) {
  int i = blockIdx.x * blockDim.x + threadIdx.x;
  if (i < N_NODES * 8) {
    int node = i >> 3, j = i & 7;
    vc[i] = (j < 7) ? v[node * 7 + j] : labels[node];
  }
}

// goal = vc[argmax(labels)]
__global__ void k_goal(const float* labels, const float* vc, float* goal) {
  if (blockIdx.x == 0 && threadIdx.x == 0) {
    int gi = 0; float best = labels[0];
    for (int i = 1; i < N_NODES; ++i) { float l = labels[i]; if (l > best) { best = l; gi = i; } }
    for (int j = 0; j < 8; ++j) goal[j] = vc[gi * 8 + j];
  }
}

// Convert W [K=KT*32][64] f32 row-major into f16 WMMA B-fragment order:
// frag[kt][nt][lane][e] = W[kt*32 + (lane<16?0:8) + (e<8 ? e : e+8)][nt*16 + (lane&15)]
__global__ void k_wfrag(const float* W, _Float16* frag, int KT) {
  int idx = blockIdx.x * blockDim.x + threadIdx.x;
  if (idx >= KT * 2048) return;
  int e  = idx & 15;
  int l  = (idx >> 4) & 31;
  int nt = (idx >> 9) & 3;
  int kt = idx >> 11;
  int k = kt * 32 + ((l < 16) ? 0 : 8) + ((e < 8) ? e : e + 8);
  int n = nt * 16 + (l & 15);
  frag[idx] = (_Float16)W[k * 64 + n];
}

// x = MLP_hx([vc, goal, d, d*d]) : per node, 64-thread block
__global__ __launch_bounds__(64) void k_init_x(const float* vc, const float* goal,
                                               const float* W1, const float* b1,
                                               const float* W2, const float* b2,
                                               float* x) {
  __shared__ float sin_[32];
  __shared__ float sh[64];
  int i = blockIdx.x, t = threadIdx.x;
  if (t < 8) {
    float vcv = vc[i * 8 + t], g = goal[t], d = vcv - g;
    sin_[t] = vcv; sin_[8 + t] = g; sin_[16 + t] = d; sin_[24 + t] = d * d;
  }
  __syncthreads();
  float acc = b1[t];
  for (int k = 0; k < 32; ++k) acc += sin_[k] * W1[k * 64 + t];
  sh[t] = acc > 0.f ? acc : 0.f;
  __syncthreads();
  float o = b2[t];
  for (int h = 0; h < 64; ++h) o += sh[h] * W2[h * 64 + t];
  x[i * 64 + t] = o;
}

// y = MLP_hy([vj - vi, vj, vi]) : per edge, 64-thread block
__global__ __launch_bounds__(64) void k_init_y(const float* vc, const int* src, const int* tgt,
                                               const float* W1, const float* b1,
                                               const float* W2, const float* b2,
                                               float* y) {
  __shared__ float sin_[24];
  __shared__ float sh[64];
  int e = blockIdx.x, t = threadIdx.x;
  int s = src[e], g = tgt[e];
  if (t < 8) {
    float vi = vc[s * 8 + t], vj = vc[g * 8 + t];
    sin_[t] = vj - vi; sin_[8 + t] = vj; sin_[16 + t] = vi;
  }
  __syncthreads();
  float acc = b1[t];
  for (int k = 0; k < 24; ++k) acc += sin_[k] * W1[k * 64 + t];
  sh[t] = acc > 0.f ? acc : 0.f;
  __syncthreads();
  float o = b2[t];
  for (int h = 0; h < 64; ++h) o += sh[h] * W2[h * 64 + t];
  y[e * 64 + t] = o;
}

__global__ void k_init_agg(float* agg) {
  int i = blockIdx.x * blockDim.x + threadIdx.x;
  if (i < N_NODES * 64) agg[i] = -__builtin_inff();
}

// -------------------------------------------------------------- WMMA loop
// fx: msg = MLP_fx([xj-xi, xj, xi, y]) (256 -> 64 -> 64), atomic segment max on tgt.
// Block = 128 threads (4 waves); each wave owns a 16-edge tile; block covers 64 edges.
__global__ __launch_bounds__(128) void k_fx(const float* __restrict__ x,
                                            const float* __restrict__ y,
                                            float* __restrict__ agg,
                                            const int* __restrict__ src,
                                            const int* __restrict__ tgt,
                                            const _Float16* __restrict__ W1f, const float* __restrict__ b1,
                                            const _Float16* __restrict__ W2f, const float* __restrict__ b2) {
  __shared__ _Float16 s_in[64][256];  // 32 KB
  __shared__ _Float16 s_h[64][64];    //  8 KB
  int t = threadIdx.x;
  int blockEdge0 = blockIdx.x * 64;

  // stage inputs (coalesced on feature)
  int f = t & 63;
  for (int el = (t >> 6); el < 64; el += 2) {
    int e = blockEdge0 + el;
    float xj = x[src[e] * 64 + f];   // x_j = x[source]
    float xi = x[tgt[e] * 64 + f];   // x_i = x[target]
    float yv = y[e * 64 + f];
    s_in[el][f]       = (_Float16)(xj - xi);
    s_in[el][64 + f]  = (_Float16)xj;
    s_in[el][128 + f] = (_Float16)xi;
    s_in[el][192 + f] = (_Float16)yv;
  }
  __syncthreads();

  int wave = t >> 5, lane = t & 31;
  int row = lane & 15, hi = lane >> 4;
  int edgeBase = wave * 16;

  // layer 1: 256 -> 64, 8 K-tiles x 4 N-tiles
  v8f acc[4];
#pragma unroll
  for (int nt = 0; nt < 4; ++nt) {
    float bv = b1[nt * 16 + row];
    v8f tv;
#pragma unroll
    for (int r = 0; r < 8; ++r) tv[r] = bv;
    acc[nt] = tv;
  }
#pragma unroll
  for (int kt = 0; kt < 8; ++kt) {
    int K0 = kt * 32 + hi * 8;
    const _Float16* base = &s_in[edgeBase + row][0];
    v8h alo = *(const v8h*)(base + K0);
    v8h ahi = *(const v8h*)(base + K0 + 16);
    v16h a = __builtin_shufflevector(alo, ahi, 0,1,2,3,4,5,6,7,8,9,10,11,12,13,14,15);
#pragma unroll
    for (int nt = 0; nt < 4; ++nt) {
      v16h b = *(const v16h*)(W1f + (((kt * 4 + nt) * 32 + lane) << 4));
      acc[nt] = __builtin_amdgcn_wmma_f32_16x16x32_f16(false, a, false, b, (short)0,
                                                       acc[nt], false, false);
    }
  }
  // relu, stash hidden to LDS as f16 (D layout: m = r + 8*hi, n = nt*16 + row)
#pragma unroll
  for (int nt = 0; nt < 4; ++nt)
#pragma unroll
    for (int r = 0; r < 8; ++r) {
      float h = acc[nt][r];
      s_h[edgeBase + r + 8 * hi][nt * 16 + row] = (_Float16)(h > 0.f ? h : 0.f);
    }
  __syncthreads();

  // layer 2: 64 -> 64, 2 K-tiles x 4 N-tiles
  v8f acc2[4];
#pragma unroll
  for (int nt = 0; nt < 4; ++nt) {
    float bv = b2[nt * 16 + row];
    v8f tv;
#pragma unroll
    for (int r = 0; r < 8; ++r) tv[r] = bv;
    acc2[nt] = tv;
  }
#pragma unroll
  for (int kt = 0; kt < 2; ++kt) {
    int K0 = kt * 32 + hi * 8;
    const _Float16* base = &s_h[edgeBase + row][0];
    v8h alo = *(const v8h*)(base + K0);
    v8h ahi = *(const v8h*)(base + K0 + 16);
    v16h a = __builtin_shufflevector(alo, ahi, 0,1,2,3,4,5,6,7,8,9,10,11,12,13,14,15);
#pragma unroll
    for (int nt = 0; nt < 4; ++nt) {
      v16h b = *(const v16h*)(W2f + (((kt * 4 + nt) * 32 + lane) << 4));
      acc2[nt] = __builtin_amdgcn_wmma_f32_16x16x32_f16(false, a, false, b, (short)0,
                                                        acc2[nt], false, false);
    }
  }
  // segment max onto target nodes
#pragma unroll
  for (int nt = 0; nt < 4; ++nt)
#pragma unroll
    for (int r = 0; r < 8; ++r) {
      int e = blockEdge0 + edgeBase + r + 8 * hi;
      int n = nt * 16 + row;
      atomicMaxF(&agg[tgt[e] * 64 + n], acc2[nt][r]);
    }
}

// x = max(x, where(agg==-inf, 0, agg)); re-arm agg for next iteration
__global__ void k_combine(float* x, float* agg) {
  int i = blockIdx.x * blockDim.x + threadIdx.x;
  if (i < N_NODES * 64) {
    float a = agg[i];
    if (a == -__builtin_inff()) a = 0.f;
    float xv = x[i];
    x[i] = xv > a ? xv : a;
    agg[i] = -__builtin_inff();
  }
}

// fy: y = max(y, MLP_fy([xj2-xi2, xj2, xi2])) with xi2=x[src], xj2=x[tgt] (192 -> 64 -> 64)
__global__ __launch_bounds__(128) void k_fy(const float* __restrict__ x,
                                            float* __restrict__ y,
                                            const int* __restrict__ src,
                                            const int* __restrict__ tgt,
                                            const _Float16* __restrict__ W1f, const float* __restrict__ b1,
                                            const _Float16* __restrict__ W2f, const float* __restrict__ b2) {
  __shared__ _Float16 s_in[64][192];  // 24 KB
  __shared__ _Float16 s_h[64][64];    //  8 KB
  int t = threadIdx.x;
  int blockEdge0 = blockIdx.x * 64;

  int f = t & 63;
  for (int el = (t >> 6); el < 64; el += 2) {
    int e = blockEdge0 + el;
    float xi = x[src[e] * 64 + f];   // xi2 = x[src]
    float xj = x[tgt[e] * 64 + f];   // xj2 = x[tgt]
    s_in[el][f]       = (_Float16)(xj - xi);
    s_in[el][64 + f]  = (_Float16)xj;
    s_in[el][128 + f] = (_Float16)xi;
  }
  __syncthreads();

  int wave = t >> 5, lane = t & 31;
  int row = lane & 15, hi = lane >> 4;
  int edgeBase = wave * 16;

  v8f acc[4];
#pragma unroll
  for (int nt = 0; nt < 4; ++nt) {
    float bv = b1[nt * 16 + row];
    v8f tv;
#pragma unroll
    for (int r = 0; r < 8; ++r) tv[r] = bv;
    acc[nt] = tv;
  }
#pragma unroll
  for (int kt = 0; kt < 6; ++kt) {
    int K0 = kt * 32 + hi * 8;
    const _Float16* base = &s_in[edgeBase + row][0];
    v8h alo = *(const v8h*)(base + K0);
    v8h ahi = *(const v8h*)(base + K0 + 16);
    v16h a = __builtin_shufflevector(alo, ahi, 0,1,2,3,4,5,6,7,8,9,10,11,12,13,14,15);
#pragma unroll
    for (int nt = 0; nt < 4; ++nt) {
      v16h b = *(const v16h*)(W1f + (((kt * 4 + nt) * 32 + lane) << 4));
      acc[nt] = __builtin_amdgcn_wmma_f32_16x16x32_f16(false, a, false, b, (short)0,
                                                       acc[nt], false, false);
    }
  }
#pragma unroll
  for (int nt = 0; nt < 4; ++nt)
#pragma unroll
    for (int r = 0; r < 8; ++r) {
      float h = acc[nt][r];
      s_h[edgeBase + r + 8 * hi][nt * 16 + row] = (_Float16)(h > 0.f ? h : 0.f);
    }
  __syncthreads();

  v8f acc2[4];
#pragma unroll
  for (int nt = 0; nt < 4; ++nt) {
    float bv = b2[nt * 16 + row];
    v8f tv;
#pragma unroll
    for (int r = 0; r < 8; ++r) tv[r] = bv;
    acc2[nt] = tv;
  }
#pragma unroll
  for (int kt = 0; kt < 2; ++kt) {
    int K0 = kt * 32 + hi * 8;
    const _Float16* base = &s_h[edgeBase + row][0];
    v8h alo = *(const v8h*)(base + K0);
    v8h ahi = *(const v8h*)(base + K0 + 16);
    v16h a = __builtin_shufflevector(alo, ahi, 0,1,2,3,4,5,6,7,8,9,10,11,12,13,14,15);
#pragma unroll
    for (int nt = 0; nt < 4; ++nt) {
      v16h b = *(const v16h*)(W2f + (((kt * 4 + nt) * 32 + lane) << 4));
      acc2[nt] = __builtin_amdgcn_wmma_f32_16x16x32_f16(false, a, false, b, (short)0,
                                                        acc2[nt], false, false);
    }
  }
#pragma unroll
  for (int nt = 0; nt < 4; ++nt)
#pragma unroll
    for (int r = 0; r < 8; ++r) {
      int e = blockEdge0 + edgeBase + r + 8 * hi;
      int n = nt * 16 + row;
      float yo = y[e * 64 + n];
      float v  = acc2[nt][r];
      y[e * 64 + n] = yo > v ? yo : v;
    }
}

// ---------------------------------------------------------------- epilogue
__global__ void k_zero(float* out, long long n) {
  long long stride = (long long)gridDim.x * blockDim.x;
  for (long long i = (long long)blockIdx.x * blockDim.x + threadIdx.x; i < n; i += stride)
    out[i] = 0.0f;
}

__global__ void k_scatter(const float* y, const int* src, const int* tgt, float* out) {
  int idx = blockIdx.x * blockDim.x + threadIdx.x;
  if (idx < N_EDGES * 64) {
    int e = idx >> 6, n = idx & 63;
    out[((long long)src[e] * N_NODES + tgt[e]) * 64 + n] = y[idx];
  }
}

__global__ void k_copy_x(const float* x, float* out) {
  int i = blockIdx.x * blockDim.x + threadIdx.x;
  if (i < N_NODES * 64) out[(long long)N_NODES * N_NODES * 64 + i] = x[i];
}

// ------------------------------------------------------------------- host
extern "C" void kernel_launch(void* const* d_in, const int* in_sizes, int n_in,
                              void* d_out, int out_size, void* d_ws, size_t ws_size,
                              hipStream_t stream) {
  const float* v        = (const float*)d_in[0];
  const float* labels   = (const float*)d_in[1];
  const int*   edge_idx = (const int*)d_in[4];
  const int*   src = edge_idx;
  const int*   tgt = edge_idx + N_EDGES;
  const float* hx_W1 = (const float*)d_in[6],  *hx_b1 = (const float*)d_in[7];
  const float* hx_W2 = (const float*)d_in[8],  *hx_b2 = (const float*)d_in[9];
  const float* hy_W1 = (const float*)d_in[10], *hy_b1 = (const float*)d_in[11];
  const float* hy_W2 = (const float*)d_in[12], *hy_b2 = (const float*)d_in[13];
  const float* fx_W1 = (const float*)d_in[14], *fx_b1 = (const float*)d_in[15];
  const float* fx_W2 = (const float*)d_in[16], *fx_b2 = (const float*)d_in[17];
  const float* fy_W1 = (const float*)d_in[18], *fy_b1 = (const float*)d_in[19];
  const float* fy_W2 = (const float*)d_in[20], *fy_b2 = (const float*)d_in[21];
  float* out = (float*)d_out;

  // workspace layout
  float* x    = (float*)d_ws;                    // 96000
  float* y    = x + N_NODES * 64;                // 1,536,000
  float* agg  = y + N_EDGES * 64;                // 96000
  float* vc   = agg + N_NODES * 64;              // 12000
  float* goal = vc + N_NODES * 8;                // 8
  _Float16* fxW1f = (_Float16*)(goal + 8);       // 8*2048 halves
  _Float16* fxW2f = fxW1f + 8 * 2048;            // 2*2048
  _Float16* fyW1f = fxW2f + 2 * 2048;            // 6*2048
  _Float16* fyW2f = fyW1f + 6 * 2048;            // 2*2048

  // prep
  k_vc<<<(N_NODES * 8 + 255) / 256, 256, 0, stream>>>(v, labels, vc);
  k_goal<<<1, 32, 0, stream>>>(labels, vc, goal);
  k_wfrag<<<(8 * 2048 + 255) / 256, 256, 0, stream>>>(fx_W1, fxW1f, 8);
  k_wfrag<<<(2 * 2048 + 255) / 256, 256, 0, stream>>>(fx_W2, fxW2f, 2);
  k_wfrag<<<(6 * 2048 + 255) / 256, 256, 0, stream>>>(fy_W1, fyW1f, 6);
  k_wfrag<<<(2 * 2048 + 255) / 256, 256, 0, stream>>>(fy_W2, fyW2f, 2);
  k_init_x<<<N_NODES, 64, 0, stream>>>(vc, goal, hx_W1, hx_b1, hx_W2, hx_b2, x);
  k_init_y<<<N_EDGES, 64, 0, stream>>>(vc, src, tgt, hy_W1, hy_b1, hy_W2, hy_b2, y);
  k_init_agg<<<(N_NODES * 64 + 255) / 256, 256, 0, stream>>>(agg);

  // message-passing loop (NE = 24000 = 375 blocks * 64 edges exactly)
  for (int it = 0; it < LOOP_N; ++it) {
    k_fx<<<N_EDGES / 64, 128, 0, stream>>>(x, y, agg, src, tgt, fxW1f, fx_b1, fxW2f, fx_b2);
    k_combine<<<(N_NODES * 64 + 255) / 256, 256, 0, stream>>>(x, agg);
    k_fy<<<N_EDGES / 64, 128, 0, stream>>>(x, y, src, tgt, fyW1f, fy_b1, fyW2f, fy_b2);
  }

  // outputs: edge_feat [N,N,64] then x [N,64]
  long long total = (long long)out_size;
  k_zero<<<2048, 256, 0, stream>>>(out, total);
  k_scatter<<<(N_EDGES * 64 + 255) / 256, 256, 0, stream>>>(y, src, tgt, out);
  k_copy_x<<<(N_NODES * 64 + 255) / 256, 256, 0, stream>>>(x, out);
}
